// Model_56968446214860
// MI455X (gfx1250) — compile-verified, matrix operand order checked by reading, not run on hardware
//
#include <hip/hip_runtime.h>

// ---------------- problem constants (match reference setup) ----------------
#define NW        1024      // num_words
#define DIM       768       // hidden size
#define NS        8192      // num spans
#define MAXW      30        // MAX_SPAN_WIDTH
#define FEAT      20        // feature size
#define SPAN_DIM  2324      // 3*768+20
#define K1PAD     2336      // SPAN_DIM padded to 32
#define FFNN      1000
#define NPAD      1024      // FFNN padded

typedef __bf16 v8bf  __attribute__((ext_vector_type(8)));
typedef __bf16 v16bf __attribute__((ext_vector_type(16)));
typedef float  v8f   __attribute__((ext_vector_type(8)));

__device__ __forceinline__ unsigned short f2bf(float f) {
    unsigned int u = __float_as_uint(f);
    u = (u + 0x7FFFu + ((u >> 16) & 1u)) >> 16;   // round-to-nearest-even
    return (unsigned short)u;
}

__device__ __forceinline__ v16bf ld_frag(const unsigned short* p) {
    v8bf lo = *reinterpret_cast<const v8bf*>(p);        // K 0..7   (16B)
    v8bf hi = *reinterpret_cast<const v8bf*>(p + 16);   // K 16..23 (16B)
    return __builtin_shufflevector(lo, hi, 0,1,2,3,4,5,6,7,8,9,10,11,12,13,14,15);
}

__device__ __forceinline__ v8f wmma_bf16(v16bf a, v16bf b, v8f c) {
    return __builtin_amdgcn_wmma_f32_16x16x32_bf16(false, a, false, b,
                                                   (short)0, c, false, false);
}

// ---------------- 1) head scores: doc[NW,DIM] @ w_head + b_head -----------
__global__ __launch_bounds__(256)
void head_scores_kernel(const float* __restrict__ doc,
                        const float* __restrict__ w_head,
                        const float* __restrict__ b_head,
                        float* __restrict__ head_scores) {
    int row  = blockIdx.x * 8 + (threadIdx.x >> 5);
    int lane = threadIdx.x & 31;
    float acc = 0.f;
    const float* r = doc + (size_t)row * DIM;
    for (int d = lane; d < DIM; d += 32) acc += r[d] * w_head[d];
    for (int off = 16; off; off >>= 1) acc += __shfl_xor(acc, off, 32);
    if (lane == 0) head_scores[row] = acc + b_head[0];
}

// ---------------- 2) pack weights: W[K,N] f32 -> Wt[Npad,Kpad] bf16 -------
__global__ __launch_bounds__(256)
void pack_wt_kernel(const float* __restrict__ W, unsigned short* __restrict__ Wt,
                    int K, int N, int Kpad, int Npad) {
    int total = Npad * Kpad;
    for (int i = blockIdx.x * blockDim.x + threadIdx.x; i < total;
         i += gridDim.x * blockDim.x) {
        int n = i / Kpad, k = i - n * Kpad;
        float v = (k < K && n < N) ? W[(size_t)k * N + n] : 0.f;
        Wt[i] = f2bf(v);
    }
}

// ---------------- 3) span embedding (softmax attention) -> bf16 rows ------
__global__ __launch_bounds__(256)
void span_emb_kernel(const float* __restrict__ doc,
                     const float* __restrict__ width_tab,
                     const float* __restrict__ head_scores,
                     const int* __restrict__ starts,
                     const int* __restrict__ ends,
                     unsigned short* __restrict__ span_emb) {
    __shared__ float sh_attn[32];
    int span = blockIdx.x;
    int s = starts[span], e = ends[span];
    int w = e - s + 1;
    if (threadIdx.x < 32) {
        int lane = threadIdx.x;
        int idx  = min(s + lane, NW - 1);
        float hs = (lane < w) ? head_scores[idx] : -1e30f;
        float m  = hs;
        for (int off = 16; off; off >>= 1) m = fmaxf(m, __shfl_xor(m, off, 32));
        float ev = (lane < w) ? __expf(hs - m) : 0.f;
        float sum = ev;
        for (int off = 16; off; off >>= 1) sum += __shfl_xor(sum, off, 32);
        sh_attn[lane] = ev / sum;
    }
    __syncthreads();
    unsigned short* row = span_emb + (size_t)span * K1PAD;
    for (int d = threadIdx.x; d < DIM; d += 256) {
        float acc = 0.f;
        for (int j = 0; j < w; ++j) {
            int idx = min(s + j, NW - 1);
            acc += sh_attn[j] * doc[(size_t)idx * DIM + d];
        }
        row[d]                  = f2bf(doc[(size_t)s * DIM + d]);  // start emb
        row[DIM + d]            = f2bf(doc[(size_t)e * DIM + d]);  // end emb
        row[2 * DIM + FEAT + d] = f2bf(acc);                       // attended
    }
    if (threadIdx.x < FEAT)
        row[2 * DIM + threadIdx.x] = f2bf(width_tab[(size_t)(w - 1) * FEAT + threadIdx.x]);
    if (threadIdx.x < K1PAD - SPAN_DIM)
        row[SPAN_DIM + threadIdx.x] = 0;      // K padding
}

// ---------------- 4) bf16 WMMA GEMM: out = relu(A @ Bt^T + bias) ----------
// A [M,K] bf16 row-major, Bt [N,K] bf16 (i.e. B column-major), tile 128x128,
// 8 waves, wave tile 32x64 (2x4 of 16x16 wmma).
__global__ __launch_bounds__(256)
void gemm_bf16_wmma(const unsigned short* __restrict__ A,
                    const unsigned short* __restrict__ Bt,
                    const float* __restrict__ bias, int biasN,
                    float* __restrict__ outF, unsigned short* __restrict__ outH,
                    int M, int N, int K) {
    int lane = threadIdx.x & 31;
    int wave = threadIdx.x >> 5;
    int wm = wave & 3, wn = wave >> 2;
    int m0 = blockIdx.y * 128 + wm * 32;
    int n0 = blockIdx.x * 128 + wn * 64;

    int mA0 = m0 + (lane & 15);
    int mA1 = mA0 + 16;
    int klane = (lane >> 4) << 3;          // lanes 0-15: K{0..7,16..23}; 16-31: +8

    v8f c[2][4];
#pragma unroll
    for (int i = 0; i < 2; ++i)
#pragma unroll
        for (int j = 0; j < 4; ++j) c[i][j] = (v8f){0,0,0,0,0,0,0,0};

    const unsigned short* aP0 = A + (size_t)mA0 * K + klane;
    const unsigned short* aP1 = A + (size_t)mA1 * K + klane;
    const unsigned short* bP[4];
#pragma unroll
    for (int j = 0; j < 4; ++j)
        bP[j] = Bt + (size_t)(n0 + j * 16 + (lane & 15)) * K + klane;

    for (int k0 = 0; k0 < K; k0 += 32) {
        v16bf a0 = ld_frag(aP0 + k0);
        v16bf a1 = ld_frag(aP1 + k0);
        __builtin_prefetch(aP0 + k0 + 128, 0, 1);   // global_prefetch_b8
        v16bf b0 = ld_frag(bP[0] + k0);
        v16bf b1 = ld_frag(bP[1] + k0);
        v16bf b2 = ld_frag(bP[2] + k0);
        v16bf b3 = ld_frag(bP[3] + k0);
        c[0][0] = wmma_bf16(a0, b0, c[0][0]);
        c[0][1] = wmma_bf16(a0, b1, c[0][1]);
        c[0][2] = wmma_bf16(a0, b2, c[0][2]);
        c[0][3] = wmma_bf16(a0, b3, c[0][3]);
        c[1][0] = wmma_bf16(a1, b0, c[1][0]);
        c[1][1] = wmma_bf16(a1, b1, c[1][1]);
        c[1][2] = wmma_bf16(a1, b2, c[1][2]);
        c[1][3] = wmma_bf16(a1, b3, c[1][3]);
    }

    int rowOff = (lane >> 4) << 3;         // C/D layout: lanes 16-31 hold M=8..15
#pragma unroll
    for (int i = 0; i < 2; ++i) {
#pragma unroll
        for (int j = 0; j < 4; ++j) {
            int col = n0 + j * 16 + (lane & 15);
            float bv = (col < biasN) ? bias[col] : 0.f;
            int rbase = m0 + i * 16 + rowOff;
#pragma unroll
            for (int r = 0; r < 8; ++r) {
                float v = c[i][j][r] + bv;
                v = v > 0.f ? v : 0.f;
                size_t off = (size_t)(rbase + r) * N + col;
                if (outH) outH[off] = f2bf(v);
                else      outF[off] = v;
            }
        }
    }
}

// ---------------- 5) scores = h2[:, :FFNN] @ W3 + b3 ----------------------
__global__ __launch_bounds__(256)
void score_gemv_kernel(const float* __restrict__ h2,
                       const float* __restrict__ W3,
                       const float* __restrict__ b3,
                       float* __restrict__ scores) {
    int span = blockIdx.x * 8 + (threadIdx.x >> 5);
    int lane = threadIdx.x & 31;
    const float* r = h2 + (size_t)span * NPAD;
    float acc = 0.f;
    for (int k = lane; k < FFNN; k += 32) acc += r[k] * W3[k];
    for (int off = 16; off; off >>= 1) acc += __shfl_xor(acc, off, 32);
    if (lane == 0) scores[span] = acc + b3[0];
}

// ---------------- 6) bitonic sort of (score,idx), descending --------------
__global__ __launch_bounds__(1024)
void sort_kernel(const float* __restrict__ scores,
                 unsigned long long* __restrict__ keys_out) {
    extern __shared__ unsigned long long keys[];
    const int n = NS, nt = 1024;
    int tid = threadIdx.x;
    for (int i = tid; i < n; i += nt) {
        unsigned int u = __float_as_uint(scores[i]);
        u ^= (u & 0x80000000u) ? 0xFFFFFFFFu : 0x80000000u;  // ascending-sortable
        u = ~u;                                              // descending
        keys[i] = ((unsigned long long)u << 32) | (unsigned int)i;
    }
    __syncthreads();
    for (int k = 2; k <= n; k <<= 1) {
        for (int j = k >> 1; j > 0; j >>= 1) {
            for (int i = tid; i < n; i += nt) {
                int ixj = i ^ j;
                if (ixj > i) {
                    bool up = ((i & k) == 0);
                    unsigned long long a = keys[i], b = keys[ixj];
                    if ((a > b) == up) { keys[i] = b; keys[ixj] = a; }
                }
            }
            __syncthreads();
        }
    }
    for (int i = tid; i < n; i += nt) keys_out[i] = keys[i];
}

// ---------------- 7) greedy NMS extraction (single wave32) ----------------
__global__ __launch_bounds__(32)
void extract_kernel(const unsigned long long* __restrict__ order,
                    const int* __restrict__ starts,
                    const int* __restrict__ ends,
                    const int* __restrict__ top_ptr,
                    float* __restrict__ out_top) {
    __shared__ int sh_max_end[NW];
    __shared__ int sh_min_start[NW];
    __shared__ int sh_selidx[512];
    __shared__ unsigned long long sh_key[512];
    int lane = threadIdx.x;
    int top = min(top_ptr[0], 512);
    for (int i = lane; i < NW; i += 32) { sh_max_end[i] = -1; sh_min_start[i] = NW; }
    for (int i = lane; i < 512; i += 32) sh_selidx[i] = -1;
    __syncthreads();

    int count = 0;
    for (int i = 0; i < NS; ++i) {
        int idx = (int)(order[i] & 0xFFFFFFFFull);
        int s = starts[idx], e = ends[idx];
        bool dup = (sh_max_end[s] == e);
        int p1 = s + 1 + lane;
        bool c1 = (p1 <= e) && (sh_max_end[p1] > e);
        int p2 = s + lane;
        bool c2 = (p2 < e) && (sh_min_start[p2] < s);
        bool cross = __any(c1 || c2);
        bool accept = (count < top) && !dup && !cross;
        if (accept) {
            if (lane == 0) {
                sh_max_end[s]   = max(sh_max_end[s], e);
                sh_min_start[e] = min(sh_min_start[e], s);
                sh_selidx[count] = idx;
            }
            count++;
        }
        __syncthreads();
    }

    // sort selected by (start*NW + end); -1 slots to the back
    for (int i = lane; i < 512; i += 32) {
        int sid = (i < count) ? sh_selidx[i] : -1;
        unsigned int key = (sid >= 0)
            ? (unsigned int)(starts[sid] * NW + ends[sid]) : 0x7FFFFFFFu;
        sh_key[i] = ((unsigned long long)key << 32) | (unsigned int)sid;
    }
    __syncthreads();
    for (int k = 2; k <= 512; k <<= 1) {
        for (int j = k >> 1; j > 0; j >>= 1) {
            for (int i = lane; i < 512; i += 32) {
                int ixj = i ^ j;
                if (ixj > i) {
                    bool up = ((i & k) == 0);
                    unsigned long long a = sh_key[i], b = sh_key[ixj];
                    if ((a > b) == up) { sh_key[i] = b; sh_key[ixj] = a; }
                }
            }
            __syncthreads();
        }
    }
    for (int i = lane; i < top; i += 32)
        out_top[i] = (float)(int)(sh_key[i] & 0xFFFFFFFFull);
}

// ---------------- launch ---------------------------------------------------
extern "C" void kernel_launch(void* const* d_in, const int* in_sizes, int n_in,
                              void* d_out, int out_size, void* d_ws, size_t ws_size,
                              hipStream_t stream) {
    const float* doc    = (const float*)d_in[0];
    const float* wwidth = (const float*)d_in[1];
    const float* whead  = (const float*)d_in[2];
    const float* bhead  = (const float*)d_in[3];
    const float* W1     = (const float*)d_in[4];
    const float* b1     = (const float*)d_in[5];
    const float* W2     = (const float*)d_in[6];
    const float* b2     = (const float*)d_in[7];
    const float* W3     = (const float*)d_in[8];
    const float* b3     = (const float*)d_in[9];
    const int* starts   = (const int*)d_in[10];
    const int* ends     = (const int*)d_in[11];
    const int* topnum   = (const int*)d_in[12];

    char* ws = (char*)d_ws;
    size_t off = 0;
    auto alloc = [&](size_t bytes) { void* p = ws + off; off = (off + bytes + 255) & ~(size_t)255; return p; };
    float*              hsc   = (float*)             alloc((size_t)NW * 4);
    unsigned short*     sEmb  = (unsigned short*)    alloc((size_t)NS * K1PAD * 2);
    unsigned short*     W1t   = (unsigned short*)    alloc((size_t)NPAD * K1PAD * 2);
    unsigned short*     W2t   = (unsigned short*)    alloc((size_t)NPAD * NPAD * 2);
    unsigned short*     h1    = (unsigned short*)    alloc((size_t)NS * NPAD * 2);
    float*              h2    = (float*)             alloc((size_t)NS * NPAD * 4);
    unsigned long long* order = (unsigned long long*)alloc((size_t)NS * 8);

    float* out_scores = (float*)d_out;
    float* out_top    = (float*)d_out + NS;

    head_scores_kernel<<<NW / 8, 256, 0, stream>>>(doc, whead, bhead, hsc);
    pack_wt_kernel<<<(NPAD * K1PAD + 255) / 256, 256, 0, stream>>>(W1, W1t, SPAN_DIM, FFNN, K1PAD, NPAD);
    pack_wt_kernel<<<(NPAD * NPAD + 255) / 256, 256, 0, stream>>>(W2, W2t, FFNN, FFNN, NPAD, NPAD);
    span_emb_kernel<<<NS, 256, 0, stream>>>(doc, wwidth, hsc, starts, ends, sEmb);

    dim3 g1(NPAD / 128, NS / 128);
    gemm_bf16_wmma<<<g1, 256, 0, stream>>>(sEmb, W1t, b1, FFNN, nullptr, h1, NS, NPAD, K1PAD);
    gemm_bf16_wmma<<<g1, 256, 0, stream>>>(h1, W2t, b2, FFNN, h2, nullptr, NS, NPAD, NPAD);

    score_gemv_kernel<<<NS / 8, 256, 0, stream>>>(h2, W3, b3, out_scores);
    sort_kernel<<<1, 1024, NS * sizeof(unsigned long long), stream>>>(out_scores, order);
    extract_kernel<<<1, 32, 0, stream>>>(order, starts, ends, topnum, out_top);
}
